// ActorCriticTransformer_73710228734022
// MI455X (gfx1250) — matrix operand order repaired and checked by reading, not compiled, not run on hardware
//
#include <hip/hip_runtime.h>

// ---------------------------------------------------------------------------
// Types / helpers
// ---------------------------------------------------------------------------
typedef __attribute__((ext_vector_type(16))) __bf16 v16bf;
typedef __attribute__((ext_vector_type(8)))  float  v8f;
typedef __attribute__((ext_vector_type(4)))  unsigned int u32x4;
typedef __attribute__((ext_vector_type(4)))  int i32x4;
typedef __attribute__((ext_vector_type(8)))  int i32x8;
typedef unsigned short u16;

union Frag { v16bf v; u32x4 q[2]; u16 u[16]; };

__device__ __forceinline__ u16 f2bf(float f) {
  unsigned u = __float_as_uint(f);
  u += 0x7FFFu + ((u >> 16) & 1u);          // round-to-nearest-even
  return (u16)(u >> 16);
}

__device__ __forceinline__ float gelu_f(float x) {
  return 0.5f * x * (1.0f + erff(x * 0.70710678118654752f));
}

__device__ __forceinline__ v8f wmma_bf16(const Frag& a, const Frag& b, v8f c) {
  return __builtin_amdgcn_wmma_f32_16x16x32_bf16(false, a.v, false, b.v,
                                                 (short)0, c, false, false);
}

// TDM availability (device pass only; host pass compiles the fallback stub)
#if defined(__HIP_DEVICE_COMPILE__) && __has_builtin(__builtin_amdgcn_tensor_load_to_lds) && __has_builtin(__builtin_amdgcn_s_wait_tensorcnt)
#define ATTN_TDM 1
#else
#define ATTN_TDM 0
#endif

#if ATTN_TDM
// DMA a (rows x 64) bf16 tile (row stride 64 elements) from global into LDS.
// D# per CDNA5 ISA: group0 = count/lds_addr/global_addr/type,
// group1 = mask/data_size/dims/strides. 2D tensor -> groups 2/3 zero.
__device__ __forceinline__ void tdm_load_v(unsigned ldsOff, const u16* gptr, int rows) {
  unsigned long long ga = (unsigned long long)(size_t)gptr;
  u32x4 g0;
  g0.x = 1u;                                     // count=1, user mode
  g0.y = ldsOff;                                 // lds_addr (bytes)
  g0.z = (unsigned)ga;                           // global_addr[31:0]
  g0.w = (unsigned)((ga >> 32) & 0x01FFFFFFu)    // global_addr[56:32]
       | (2u << 30);                             // type=2 ("image")
  i32x8 g1;
  g1.s0 = 0x00010000;                            // wg_mask=0, data_size=1 (2B)
  g1.s1 = (64 & 0xFFFF) << 16;                   // tensor_dim0[15:0] = 64
  g1.s2 = (rows & 0xFFFF) << 16;                 // dim0 hi=0 | tensor_dim1 lo = rows
  g1.s3 = (64 << 16);                            // dim1 hi=0 | tile_dim0 = 64
  g1.s4 = rows & 0xFFFF;                         // tile_dim1 = rows, tile_dim2=0
  g1.s5 = 64;                                    // tensor_dim0_stride = 64
  g1.s6 = 0;
  g1.s7 = 0;
  i32x4 gz = {0, 0, 0, 0};
#if __clang_major__ >= 23
  i32x8 gz8 = {0, 0, 0, 0, 0, 0, 0, 0};
  __builtin_amdgcn_tensor_load_to_lds(g0, g1, gz, gz, gz8, 0);
#else
  __builtin_amdgcn_tensor_load_to_lds(g0, g1, gz, gz, 0);
#endif
  __builtin_amdgcn_s_wait_tensorcnt(0);
}
#endif

// Model constants: B=16 S=512 F=64 D=512 H=8 E=64 L=4 ; NT = B*S = 8192

// ---------------------------------------------------------------------------
// f32 -> bf16 cast
// ---------------------------------------------------------------------------
__global__ void cast_k(const float* __restrict__ src, u16* __restrict__ dst, int n) {
  int i = blockIdx.x * 256 + threadIdx.x;
  if (i < n) dst[i] = f2bf(src[i]);
}

// ---------------------------------------------------------------------------
// Conv stem (small fraction of FLOPs -> VALU)
// ---------------------------------------------------------------------------
__global__ void conv1_k(const float* __restrict__ x, const float* __restrict__ w,
                        const float* __restrict__ bias, float* __restrict__ out) {
  int idx = blockIdx.x * 256 + threadIdx.x;         // B*S*256
  int co = idx & 255;
  int s  = (idx >> 8) & 511;
  int b  = idx >> 17;
  float acc = bias[co];
  const float* wr = w + (size_t)co * 64 * 3;
  #pragma unroll
  for (int k = 0; k < 3; ++k) {
    int sp = s + k - 1;
    if (sp < 0 || sp >= 512) continue;
    const float* xr = x + ((size_t)b * 512 + sp) * 64;
    for (int ci = 0; ci < 64; ++ci) acc += xr[ci] * wr[ci * 3 + k];
  }
  out[idx] = gelu_f(acc);
}

__global__ void conv2_k(const float* __restrict__ h1, const float* __restrict__ w,
                        const float* __restrict__ bias, u16* __restrict__ out) {
  int idx = blockIdx.x * 256 + threadIdx.x;         // B*S*512
  int co = idx & 511;
  int s  = (idx >> 9) & 511;
  int b  = idx >> 18;
  float acc = bias[co];
  const float* wr = w + (size_t)co * 256 * 3;
  #pragma unroll
  for (int k = 0; k < 3; ++k) {
    int sp = s + k - 1;
    if (sp < 0 || sp >= 512) continue;
    const float* xr = h1 + ((size_t)b * 512 + sp) * 256;
    for (int ci = 0; ci < 256; ++ci) acc += xr[ci] * wr[ci * 3 + k];
  }
  out[idx] = f2bf(gelu_f(acc));
}

// ---------------------------------------------------------------------------
// WMMA GEMM: C[M,N] = A[M,K](bf16) @ W[N,K]^T(bf16) with fused epilogue.
// Block = 256 threads = 8 waves; each wave -> 16 rows x 64 cols (4 C tiles).
// FLAGS: 1=bias 2=gelu 4=resid(f32,M*N) 8=pos(f32,(m%512)*512+n)
//        16=f32 out row-major  32=bf16 out row-major  64=bf16 out QKV-permuted
// ---------------------------------------------------------------------------
template <int FLAGS>
__global__ __launch_bounds__(256)
void gemm_wmma_k(const u16* __restrict__ A, const u16* __restrict__ W,
                 const float* __restrict__ bias, const float* __restrict__ resid,
                 const float* __restrict__ pos,
                 float* __restrict__ outF, u16* __restrict__ outB,
                 int M, int N, int K) {
  const int lane  = threadIdx.x & 31;
  const int wav   = threadIdx.x >> 5;
  const int mBase = blockIdx.x * 128 + wav * 16;
  const int nBase = blockIdx.y * 64;
  const int half  = lane >> 4;
  const int l15   = lane & 15;

  const v8f vzero = {0.f,0.f,0.f,0.f,0.f,0.f,0.f,0.f};
  v8f acc[4] = {vzero, vzero, vzero, vzero};

  const u16* aRow = A + (size_t)(mBase + l15) * K;

  for (int kk = 0; kk < K; kk += 32) {
    Frag a;
    a.q[0] = *(const u32x4*)(aRow + kk +      half * 16);
    a.q[1] = *(const u32x4*)(aRow + kk +  8 + half * 16);
    if (kk + 64 <= K) __builtin_prefetch(aRow + kk + 64, 0, 1);
    #pragma unroll
    for (int j = 0; j < 4; ++j) {
      const int col = nBase + j * 16 + l15;
      const u32x4* wp = (const u32x4*)(W + (size_t)col * K + kk + half * 16);
      Frag b; b.q[0] = wp[0]; b.q[1] = wp[1];
      acc[j] = wmma_bf16(a, b, acc[j]);
    }
  }

  #pragma unroll
  for (int j = 0; j < 4; ++j) {
    const int col = nBase + j * 16 + l15;
    #pragma unroll
    for (int r = 0; r < 8; ++r) {
      const int row = mBase + r + 8 * half;
      float v = acc[j][r];
      if (FLAGS & 1)  v += bias[col];
      if (FLAGS & 8)  v += pos[(size_t)(row & 511) * 512 + col];
      if (FLAGS & 4)  v += resid[(size_t)row * N + col];
      if (FLAGS & 2)  v = gelu_f(v);
      if (FLAGS & 16) outF[(size_t)row * N + col] = v;
      if (FLAGS & 32) outB[(size_t)row * N + col] = f2bf(v);
      if (FLAGS & 64) {
        const int bb = row >> 9, ss = row & 511, hh = col >> 6, ee = col & 63;
        outB[(((size_t)(bb * 8 + hh) << 9) + ss) * 64 + ee] = f2bf(v);
      }
    }
  }
}

// ---------------------------------------------------------------------------
// LayerNorm over D=512, one wave per row; optional bf16 and f32 outputs.
// ---------------------------------------------------------------------------
__global__ __launch_bounds__(32)
void ln_k(const float* __restrict__ X, const float* __restrict__ g,
          const float* __restrict__ be, u16* __restrict__ outB,
          float* __restrict__ outF, int rowStride, int rowOffset) {
  const int lane = threadIdx.x;
  const int ro   = blockIdx.x;
  const float* xr = X + (size_t)(ro * rowStride + rowOffset) * 512;
  float vals[16];
  float s = 0.f;
  #pragma unroll
  for (int i = 0; i < 16; ++i) { vals[i] = xr[lane + i * 32]; s += vals[i]; }
  #pragma unroll
  for (int off = 16; off >= 1; off >>= 1) s += __shfl_xor(s, off, 32);
  const float mu = s * (1.0f / 512.0f);
  float vs = 0.f;
  #pragma unroll
  for (int i = 0; i < 16; ++i) { float d = vals[i] - mu; vs += d * d; }
  #pragma unroll
  for (int off = 16; off >= 1; off >>= 1) vs += __shfl_xor(vs, off, 32);
  const float inv = rsqrtf(vs * (1.0f / 512.0f) + 1e-5f);
  #pragma unroll
  for (int i = 0; i < 16; ++i) {
    const int c = lane + i * 32;
    const float y = (vals[i] - mu) * inv * g[c] + be[c];
    if (outB) outB[(size_t)ro * 512 + c] = f2bf(y);
    if (outF) outF[(size_t)ro * 512 + c] = y;
  }
}

// ---------------------------------------------------------------------------
// Masked attention: one wave per (b, h, 16-row tile of S).
// Q,K,V: (B,H,S,E) bf16.  O: (B*S, D) bf16 with head-concat columns.
// LDS pool (48KB): [0,16K) probs bf16; [16K,48K) scores f32, later reused as
// TDM-staged V (two 256x64 chunks).
// ---------------------------------------------------------------------------
__global__ __launch_bounds__(32)
void attn_k(const u16* __restrict__ Q, const u16* __restrict__ Kb,
            const u16* __restrict__ V, u16* __restrict__ O) {
  __shared__ __align__(16) unsigned char pool[49152];
  u16*   sP = (u16*)pool;                  // 16KB: probs 16x512 bf16
  float* sS = (float*)(pool + 16384);      // 32KB: scores 16x512 f32
  u16*   sV = (u16*)(pool + 16384);        // 32KB: staged V 256x64 bf16

  const int lane = threadIdx.x;
  const int st = blockIdx.x;   // 0..31 row tile
  const int h  = blockIdx.y;   // 0..7
  const int b  = blockIdx.z;   // 0..15
  const int half = lane >> 4, l15 = lane & 15;

  const size_t base = ((size_t)(b * 8 + h)) * 512 * 64;
  const u16* Qh = Q + base;
  const u16* Kh = Kb + base;
  const u16* Vh = V + base;

  const int hh  = h < 4 ? h : 4;
  const int dil = 1 << hh;
  const int mod = (2 << hh) - 1;

  // Q tile A-fragments for K-dim 0..31 and 32..63 (E = 64)
  Frag aq0, aq1;
  {
    const u16* qr = Qh + (size_t)(st * 16 + l15) * 64;
    aq0.q[0] = *(const u32x4*)(qr +      half * 16);
    aq0.q[1] = *(const u32x4*)(qr +  8 + half * 16);
    aq1.q[0] = *(const u32x4*)(qr + 32 + half * 16);
    aq1.q[1] = *(const u32x4*)(qr + 40 + half * 16);
  }

  const v8f vzero = {0.f,0.f,0.f,0.f,0.f,0.f,0.f,0.f};

  // ---- scores = mask(QK^T / sqrt(E)) into LDS ----
  for (int tt = 0; tt < 32; ++tt) {
    v8f acc = vzero;
    const u16* kr = Kh + (size_t)(tt * 16 + l15) * 64;
    Frag b0, b1;
    b0.q[0] = *(const u32x4*)(kr +      half * 16);
    b0.q[1] = *(const u32x4*)(kr +  8 + half * 16);
    b1.q[0] = *(const u32x4*)(kr + 32 + half * 16);
    b1.q[1] = *(const u32x4*)(kr + 40 + half * 16);
    acc = wmma_bf16(aq0, b0, acc);
    acc = wmma_bf16(aq1, b1, acc);
    const int j = tt * 16 + l15;
    #pragma unroll
    for (int r = 0; r < 8; ++r) {
      const int mloc = r + 8 * half;
      const int i = st * 16 + mloc;
      int d = i - j; d = d < 0 ? -d : d;
      const bool ok = (d <= dil) || ((d & mod) == 0);
      sS[mloc * 512 + j] = ok ? acc[r] * 0.125f : -1e30f;
    }
  }

  // ---- softmax rows (wave-cooperative, shfl reductions) ----
  for (int r = 0; r < 16; ++r) {
    float mx = -1e30f;
    for (int c = lane; c < 512; c += 32) mx = fmaxf(mx, sS[r * 512 + c]);
    #pragma unroll
    for (int off = 16; off >= 1; off >>= 1) mx = fmaxf(mx, __shfl_xor(mx, off, 32));
    float sm = 0.f;
    for (int c = lane; c < 512; c += 32) {
      float e = __expf(sS[r * 512 + c] - mx);
      sS[r * 512 + c] = e;
      sm += e;
    }
    #pragma unroll
    for (int off = 16; off >= 1; off >>= 1) sm += __shfl_xor(sm, off, 32);
    const float inv = 1.0f / sm;
    for (int c = lane; c < 512; c += 32) sP[r * 512 + c] = f2bf(sS[r * 512 + c] * inv);
  }

  // ---- O = P @ V ----
#if ATTN_TDM
  {
    v8f acc[4] = {vzero, vzero, vzero, vzero};
    const unsigned sVoff = (unsigned)(size_t)(void*)sV;  // LDS byte offset
    for (int c = 0; c < 2; ++c) {
      // Prior DS reads (scores / previous V chunk) must retire before the TDM
      // engine overwrites this LDS region; TDM is unordered w.r.t. wave DS ops.
      asm volatile("s_wait_dscnt 0x0" ::: "memory");
      tdm_load_v(sVoff, Vh + (size_t)c * 256 * 64, 256);   // waits tensorcnt 0
      asm volatile("" ::: "memory");
      for (int kk2 = 0; kk2 < 256; kk2 += 32) {
        const int kk = c * 256 + kk2;
        Frag a;
        a.q[0] = *(const u32x4*)&sP[l15 * 512 + kk +      half * 16];
        a.q[1] = *(const u32x4*)&sP[l15 * 512 + kk +  8 + half * 16];
        #pragma unroll
        for (int nt = 0; nt < 4; ++nt) {
          Frag bv;
          #pragma unroll
          for (int e2 = 0; e2 < 16; ++e2)
            bv.u[e2] = sV[(size_t)(kk2 + half * 16 + e2) * 64 + nt * 16 + l15];
          acc[nt] = wmma_bf16(a, bv, acc[nt]);
        }
      }
    }
    #pragma unroll
    for (int nt = 0; nt < 4; ++nt) {
      #pragma unroll
      for (int r = 0; r < 8; ++r) {
        const int srow = st * 16 + r + 8 * half;
        O[((size_t)(b * 512 + srow)) * 512 + h * 64 + nt * 16 + l15] =
            f2bf(acc[nt][r]);
      }
    }
  }
#else
  #pragma unroll
  for (int nt = 0; nt < 4; ++nt) {
    v8f acc = vzero;
    const int col = nt * 16 + l15;
    for (int kk = 0; kk < 512; kk += 32) {
      Frag a;
      a.q[0] = *(const u32x4*)&sP[l15 * 512 + kk +      half * 16];
      a.q[1] = *(const u32x4*)&sP[l15 * 512 + kk +  8 + half * 16];
      Frag bv;
      #pragma unroll
      for (int e2 = 0; e2 < 16; ++e2)
        bv.u[e2] = Vh[(size_t)(kk + half * 16 + e2) * 64 + col];
      acc = wmma_bf16(a, bv, acc);
    }
    #pragma unroll
    for (int r = 0; r < 8; ++r) {
      const int srow = st * 16 + r + 8 * half;
      O[((size_t)(b * 512 + srow)) * 512 + h * 64 + nt * 16 + l15] = f2bf(acc[r]);
    }
  }
#endif
}

// ---------------------------------------------------------------------------
// Small head MLP layer
// ---------------------------------------------------------------------------
__global__ void head_linear_k(const float* __restrict__ in, const float* __restrict__ w,
                              const float* __restrict__ bias, float* __restrict__ out,
                              int K, int N, int doGelu) {
  const int row = blockIdx.x;
  const int j = blockIdx.y * 64 + threadIdx.x;
  if (j >= N) return;
  const float* xr = in + (size_t)row * K;
  const float* wr = w + (size_t)j * K;
  float s = bias[j];
  for (int k = 0; k < K; ++k) s += xr[k] * wr[k];
  if (doGelu) s = gelu_f(s);
  out[(size_t)row * N + j] = s;
}

// ---------------------------------------------------------------------------
// Host launch
// ---------------------------------------------------------------------------
extern "C" void kernel_launch(void* const* d_in, const int* in_sizes, int n_in,
                              void* d_out, int out_size, void* d_ws, size_t ws_size,
                              hipStream_t stream) {
  (void)in_sizes; (void)n_in; (void)out_size; (void)ws_size;
  const float* x      = (const float*)d_in[0];
  const float* conv1w = (const float*)d_in[1];
  const float* conv1b = (const float*)d_in[2];
  const float* conv2w = (const float*)d_in[3];
  const float* conv2b = (const float*)d_in[4];
  const float* embw   = (const float*)d_in[5];
  const float* embb   = (const float*)d_in[6];
  const float* pos    = (const float*)d_in[7];
  const float* Wq     = (const float*)d_in[8];
  const float* bq     = (const float*)d_in[9];
  const float* Wk     = (const float*)d_in[10];
  const float* bk     = (const float*)d_in[11];
  const float* Wv     = (const float*)d_in[12];
  const float* bv     = (const float*)d_in[13];
  const float* Wo     = (const float*)d_in[14];
  const float* bo     = (const float*)d_in[15];
  const float* ln1g   = (const float*)d_in[16];
  const float* ln1b   = (const float*)d_in[17];
  const float* ln2g   = (const float*)d_in[18];
  const float* ln2b   = (const float*)d_in[19];
  const float* W1     = (const float*)d_in[20];
  const float* b1     = (const float*)d_in[21];
  const float* W2     = (const float*)d_in[22];
  const float* b2     = (const float*)d_in[23];
  const float* lnfg   = (const float*)d_in[24];
  const float* lnfb   = (const float*)d_in[25];
  const float* a1w = (const float*)d_in[26]; const float* a1b = (const float*)d_in[27];
  const float* a2w = (const float*)d_in[28]; const float* a2b = (const float*)d_in[29];
  const float* a3w = (const float*)d_in[30]; const float* a3b = (const float*)d_in[31];
  const float* a4w = (const float*)d_in[32]; const float* a4b = (const float*)d_in[33];
  const float* c1w = (const float*)d_in[34]; const float* c1b = (const float*)d_in[35];
  const float* c2w = (const float*)d_in[36]; const float* c2b = (const float*)d_in[37];
  const float* c3w = (const float*)d_in[38]; const float* c3b = (const float*)d_in[39];
  const float* c4w = (const float*)d_in[40]; const float* c4b = (const float*)d_in[41];

  char* wsp = (char*)d_ws;
  size_t off = 0;
  auto alloc = [&](size_t bytes) -> void* {
    void* p = wsp + off;
    off += (bytes + 255) & ~(size_t)255;
    return p;
  };

  const int NT = 8192;                       // B*S tokens
  u16*   emb_bf = (u16*)alloc((size_t)262144 * 2);
  u16*   wq_bf  = (u16*)alloc((size_t)1048576 * 2);
  u16*   wk_bf  = (u16*)alloc((size_t)1048576 * 2);
  u16*   wv_bf  = (u16*)alloc((size_t)1048576 * 2);
  u16*   wo_bf  = (u16*)alloc((size_t)1048576 * 2);
  u16*   w1_bf  = (u16*)alloc((size_t)4194304 * 2);
  u16*   w2_bf  = (u16*)alloc((size_t)4194304 * 2);
  float* h1     = (float*)alloc((size_t)2097152 * 4);     // (B,S,256)
  u16*   h0_bf  = (u16*)alloc((size_t)NT * 512 * 2);
  float* z      = (float*)alloc((size_t)NT * 512 * 4);
  u16*   ln_bf  = (u16*)alloc((size_t)NT * 512 * 2);
  u16*   q_bf   = (u16*)alloc((size_t)NT * 512 * 2);
  u16*   k_bf   = (u16*)alloc((size_t)NT * 512 * 2);
  u16*   v_bf   = (u16*)alloc((size_t)NT * 512 * 2);
  u16*   o_bf   = (u16*)alloc((size_t)NT * 512 * 2);
  u16*   m1_bf  = (u16*)alloc((size_t)NT * 2048 * 2);
  float* feat   = (float*)alloc((size_t)16 * 512 * 4);
  float* t1     = (float*)alloc((size_t)16 * 128 * 4);
  float* t2     = (float*)alloc((size_t)16 * 256 * 4);
  float* t3     = (float*)alloc((size_t)16 * 64 * 4);

  // --- weight casts ---
  cast_k<<<(262144 + 255) / 256, 256, 0, stream>>>(embw, emb_bf, 262144);
  cast_k<<<(1048576 + 255) / 256, 256, 0, stream>>>(Wq, wq_bf, 1048576);
  cast_k<<<(1048576 + 255) / 256, 256, 0, stream>>>(Wk, wk_bf, 1048576);
  cast_k<<<(1048576 + 255) / 256, 256, 0, stream>>>(Wv, wv_bf, 1048576);
  cast_k<<<(1048576 + 255) / 256, 256, 0, stream>>>(Wo, wo_bf, 1048576);
  cast_k<<<(4194304 + 255) / 256, 256, 0, stream>>>(W1, w1_bf, 4194304);
  cast_k<<<(4194304 + 255) / 256, 256, 0, stream>>>(W2, w2_bf, 4194304);

  // --- conv stem ---
  conv1_k<<<2097152 / 256, 256, 0, stream>>>(x, conv1w, conv1b, h1);
  conv2_k<<<4194304 / 256, 256, 0, stream>>>(h1, conv2w, conv2b, h0_bf);

  // --- embedding: z = h0 @ emb_w^T + emb_b + pos ---
  gemm_wmma_k<1 | 8 | 16><<<dim3(64, 8), 256, 0, stream>>>(
      h0_bf, emb_bf, embb, nullptr, pos, z, nullptr, NT, 512, 512);

  // --- transformer layers ---
  for (int l = 0; l < 4; ++l) {
    ln_k<<<NT, 32, 0, stream>>>(z, ln1g + l * 512, ln1b + l * 512,
                                ln_bf, nullptr, 1, 0);
    gemm_wmma_k<1 | 64><<<dim3(64, 8), 256, 0, stream>>>(
        ln_bf, wq_bf + (size_t)l * 262144, bq + l * 512,
        nullptr, nullptr, nullptr, q_bf, NT, 512, 512);
    gemm_wmma_k<1 | 64><<<dim3(64, 8), 256, 0, stream>>>(
        ln_bf, wk_bf + (size_t)l * 262144, bk + l * 512,
        nullptr, nullptr, nullptr, k_bf, NT, 512, 512);
    gemm_wmma_k<1 | 64><<<dim3(64, 8), 256, 0, stream>>>(
        ln_bf, wv_bf + (size_t)l * 262144, bv + l * 512,
        nullptr, nullptr, nullptr, v_bf, NT, 512, 512);

    attn_k<<<dim3(32, 8, 16), 32, 0, stream>>>(q_bf, k_bf, v_bf, o_bf);

    gemm_wmma_k<1 | 4 | 16><<<dim3(64, 8), 256, 0, stream>>>(
        o_bf, wo_bf + (size_t)l * 262144, bo + l * 512,
        z, nullptr, z, nullptr, NT, 512, 512);

    ln_k<<<NT, 32, 0, stream>>>(z, ln2g + l * 512, ln2b + l * 512,
                                ln_bf, nullptr, 1, 0);
    gemm_wmma_k<1 | 2 | 32><<<dim3(64, 32), 256, 0, stream>>>(
        ln_bf, w1_bf + (size_t)l * 1048576, b1 + l * 2048,
        nullptr, nullptr, nullptr, m1_bf, NT, 2048, 512);
    gemm_wmma_k<1 | 4 | 16><<<dim3(64, 8), 256, 0, stream>>>(
        m1_bf, w2_bf + (size_t)l * 1048576, b2 + l * 512,
        z, nullptr, z, nullptr, NT, 512, 2048);
  }

  // --- final LN on last token only -> feat (16,512) ---
  ln_k<<<16, 32, 0, stream>>>(z, lnfg, lnfb, nullptr, feat, 512, 511);

  // --- heads ---
  float* outPolicy = (float*)d_out;        // (16,3)
  float* outValue  = (float*)d_out + 48;   // (16,1)
  head_linear_k<<<dim3(16, 2), 64, 0, stream>>>(feat, a1w, a1b, t1, 512, 128, 1);
  head_linear_k<<<dim3(16, 4), 64, 0, stream>>>(t1, a2w, a2b, t2, 128, 256, 1);
  head_linear_k<<<dim3(16, 1), 64, 0, stream>>>(t2, a3w, a3b, t3, 256, 57, 1);
  head_linear_k<<<dim3(16, 1), 64, 0, stream>>>(t3, a4w, a4b, outPolicy, 57, 3, 0);
  head_linear_k<<<dim3(16, 2), 64, 0, stream>>>(feat, c1w, c1b, t1, 512, 128, 1);
  head_linear_k<<<dim3(16, 4), 64, 0, stream>>>(t1, c2w, c2b, t2, 128, 256, 1);
  head_linear_k<<<dim3(16, 1), 64, 0, stream>>>(t2, c3w, c3b, t3, 256, 57, 1);
  head_linear_k<<<dim3(16, 1), 64, 0, stream>>>(t3, c4w, c4b, outValue, 57, 1, 0);
}
